// BinaryCorrelationMatcher_32512902430804
// MI455X (gfx1250) — matrix operand order repaired and checked by reading, not compiled, not run on hardware
//
#include <hip/hip_runtime.h>
#include <hip/hip_bf16.h>

// Problem constants (reference: B=4, H=480, W=640, TEMPLATE=21, SEARCH=9)
#define BB   4
#define HH   480
#define WW   640
#define RAD  10      // TEMPLATE//2
#define SRCH 9

// Tiling: one workgroup (8 wave32) = 16 rows x 128 cols of output.
#define TY 16
#define TX 128
#define NWAVE 8

// LDS geometry
#define P1R 36        // 16 + 2*RAD
#define P1C 148       // 128 + 2*RAD   (74 column pairs)
#define P2R 54        // 16 + 2*(RAD+SRCH)
#define P2C 166       // 128 + 2*(RAD+SRCH)
#define QR  48        // 36 rows valid, rows 36..47 kept zero for WMMA operand loads
#define QS  152       // col stride; cols 148..151 kept zero
#define HR  48        // H rows per column (36 valid, rest zero)
#define BSTR 72       // per-lane band-table stride in ushorts (144 B, 16B-aligned)

typedef __attribute__((ext_vector_type(16))) __bf16   v16bf;
typedef __attribute__((ext_vector_type(8)))  float    v8f;
typedef __attribute__((ext_vector_type(4)))  unsigned uv4;

union BfVec { unsigned short h[16]; uv4 q[2]; v16bf v; };

static __device__ __forceinline__ unsigned short f2bf(float f) {
  union { float f; unsigned u; } t; t.f = f;
  unsigned r = t.u + 0x7FFFu + ((t.u >> 16) & 1u);   // RNE
  return (unsigned short)(r >> 16);
}
static __device__ __forceinline__ float bf2f(unsigned short h) {
  union { unsigned u; float f; } t; t.u = ((unsigned)h) << 16;
  return t.f;
}

// Pack two f32 -> packed bf16 pair (low = a, high = b).
#if __has_builtin(__builtin_amdgcn_cvt_pk_bf16_f32)
typedef __attribute__((ext_vector_type(2))) __bf16 bf2_t;
static __device__ __forceinline__ unsigned pk2bf(float a, float b) {
  bf2_t p = __builtin_amdgcn_cvt_pk_bf16_f32(a, b);
  union { bf2_t v; unsigned u; } t; t.v = p; return t.u;
}
#else
static __device__ __forceinline__ unsigned pk2bf(float a, float b) {
  return (unsigned)f2bf(a) | ((unsigned)f2bf(b) << 16);
}
#endif

// 21x21 box-sum of the bf16 field in s_q via banded-ones WMMAs.
// Horizontal: D = A(Q rows) x B(band). Vertical: D = A(band) x B(H cols).
// Band operands are fetched from the per-lane LDS table (keeps them out of VALU).
// Returns per-lane 8 f32 outputs (C-layout: col = lane&15, rows = (lane>=16?8:0)+j).
static __device__ __forceinline__ v8f boxsum21(
    const unsigned short* qbase,   // &s_q[0][0]
    unsigned short*       hbase,   // &s_h[wave][0][0] (16 cols x HR rows, row-contig)
    const unsigned short* bl,      // &s_band[lane][0]
    int xl, int m16, int hi)
{
  BfVec vA1, vA2, vB1, vB2;
  vA1.q[0] = *(const uv4*)(bl);        vA1.q[1] = *(const uv4*)(bl + 8);
  vA2.q[0] = *(const uv4*)(bl + 16);   vA2.q[1] = *(const uv4*)(bl + 24);
  vB1.q[0] = *(const uv4*)(bl + 32);   vB1.q[1] = *(const uv4*)(bl + 40);
  vB2.q[0] = *(const uv4*)(bl + 48);   vB2.q[1] = *(const uv4*)(bl + 56);

  const int aoff = hi ? 8 : 0;     // A-layout lane K base
  #pragma unroll
  for (int blk = 0; blk < 3; ++blk) {
    const int rb = blk * 16;
    const unsigned short* qrow = qbase + (rb + m16) * QS;
    BfVec a1, a2;
    a1.q[0] = *(const uv4*)(qrow + xl + aoff);        // K = aoff+0..7
    a1.q[1] = *(const uv4*)(qrow + xl + aoff + 16);   // K = aoff+16..23
    a2.q[0] = *(const uv4*)(qrow + xl + aoff + 8);    // chunk2: cols xl+8+K
    a2.q[1] = *(const uv4*)(qrow + xl + aoff + 24);
    v8f h = {0.f,0.f,0.f,0.f,0.f,0.f,0.f,0.f};
    h = __builtin_amdgcn_wmma_f32_16x16x32_bf16(false, a1.v, false, vB1.v, (short)0, h, false, false);
    h = __builtin_amdgcn_wmma_f32_16x16x32_bf16(false, a2.v, false, vB2.v, (short)0, h, false, false);
    uv4 st;
    st[0] = pk2bf(h[0], h[1]);
    st[1] = pk2bf(h[2], h[3]);
    st[2] = pk2bf(h[4], h[5]);
    st[3] = pk2bf(h[6], h[7]);
    // H column-major: col n = m16, 8 contiguous rows starting at rb + 8*hi
    *(uv4*)(hbase + m16 * HR + rb + aoff) = st;
  }
  // Vertical pass: B operand = H column (same-wave LDS RAW -> s_wait_dscnt by compiler)
  const unsigned short* hcol = hbase + m16 * HR;
  const int boff = hi ? 16 : 0;    // B-layout lane K base
  BfVec b1, b2;
  b1.q[0] = *(const uv4*)(hcol + boff);          // rows boff+0..7
  b1.q[1] = *(const uv4*)(hcol + boff + 8);      // rows boff+8..15
  b2.q[0] = *(const uv4*)(hcol + boff + 8);      // chunk2: rows 8+boff..
  b2.q[1] = *(const uv4*)(hcol + boff + 16);
  v8f acc = {0.f,0.f,0.f,0.f,0.f,0.f,0.f,0.f};
  acc = __builtin_amdgcn_wmma_f32_16x16x32_bf16(false, vA1.v, false, b1.v, (short)0, acc, false, false);
  acc = __builtin_amdgcn_wmma_f32_16x16x32_bf16(false, vA2.v, false, b2.v, (short)0, acc, false, false);
  return acc;
}

__global__ __launch_bounds__(256, 1)
void BinaryCorrelationMatcher_kernel(const float* __restrict__ p1g,
                                     const float* __restrict__ p2g,
                                     float* __restrict__ out)
{
  __shared__ __align__(16) unsigned short s_p1[P1R][P1C];          // bf16(p1), 0xFF80 = out-of-image
  __shared__ __align__(16) unsigned short s_p2[P2R][P2C];          // bf16(p2), zero-extended
  __shared__ __align__(16) unsigned short s_q [QR][QS];            // agreement tile (bf16)
  __shared__ __align__(16) unsigned short s_h [NWAVE][16][HR];     // per-wave H scratch (bf16)
  __shared__ __align__(16) unsigned short s_band[32][BSTR];        // per-lane band operands

  const int tid  = threadIdx.x;
  const int lane = tid & 31;
  const int w    = tid >> 5;
  const int m16  = lane & 15;
  const int hi   = lane >> 4;          // lane half
  const int x0   = blockIdx.x * TX;
  const int y0   = blockIdx.y * TY;
  const int b    = blockIdx.z;
  const int xl   = w * 16;

  const float* p1b = p1g + (size_t)b * HH * WW;
  const float* p2b = p2g + (size_t)b * HH * WW;

  // ---- banded-ones operand table (built once; depends only on lane) ----
  if (tid < 32) {
    #pragma unroll
    for (int i = 0; i < 16; ++i) {
      const int kA = (hi ? 16 : 0) + i;                    // contiguous K map (matches H-column B loads)
      const int kB = (hi ? 8 : 0) + (i < 8 ? i : i + 8);   // table K map (matches Q-row A loads)
      // vertical A bands (row m = m16): taps rows m..m+20
      s_band[lane][i]      = (kA >= m16 && kA <= m16 + 20)       ? (unsigned short)0x3F80 : (unsigned short)0;
      s_band[lane][16 + i] = (kA + 8 >= 32 && kA + 8 <= m16 + 20) ? (unsigned short)0x3F80 : (unsigned short)0;
      // horizontal B bands (col n = m16): taps cols n..n+20
      s_band[lane][32 + i] = (kB >= m16 && kB <= m16 + 20)       ? (unsigned short)0x3F80 : (unsigned short)0;
      s_band[lane][48 + i] = (kB + 8 >= 32 && kB + 8 <= m16 + 20) ? (unsigned short)0x3F80 : (unsigned short)0;
    }
    #pragma unroll
    for (int i = 64; i < BSTR; ++i) s_band[lane][i] = 0;
  }

  // ---- zero Q/H once: regions read by WMMA operand loads but never written must be 0 ----
  for (int i = tid; i < QR * QS; i += 256)          ((unsigned short*)s_q)[i] = 0;
  for (int i = tid; i < NWAVE * 16 * HR; i += 256)  ((unsigned short*)s_h)[i] = 0;

  // ---- stage tiles into LDS as bf16 ----
  for (int i = tid; i < P1R * P1C; i += 256) {
    const int r = i / P1C, c = i % P1C;
    const int gy = y0 - RAD + r, gx = x0 - RAD + c;
    unsigned short v = 0xFF80;                            // sentinel: outside image
    if (gy >= 0 && gy < HH && gx >= 0 && gx < WW) v = f2bf(p1b[gy * WW + gx]);
    s_p1[r][c] = v;
  }
  for (int i = tid; i < P2R * P2C; i += 256) {
    const int r = i / P2C, c = i % P2C;
    const int gy = y0 - (RAD + SRCH) + r, gx = x0 - (RAD + SRCH) + c;
    unsigned short v = 0;                                 // zero-extended p2
    if (gy >= 0 && gy < HH && gx >= 0 && gx < WW) v = f2bf(p2b[gy * WW + gx]);
    s_p2[r][c] = v;
  }
  __syncthreads();

  const unsigned short* bl = &s_band[lane][0];

  // ---- confidence = clip(boxsum(p1)/441, 0, 1), via the same WMMA pipeline ----
  for (int i = tid; i < P1R * (P1C / 2); i += 256) {
    const int r = i / (P1C / 2), c2 = (i % (P1C / 2)) * 2;
    const unsigned u  = *(const unsigned*)&s_p1[r][c2];
    const unsigned lo = u & 0xFFFFu, hi16 = u >> 16;
    const unsigned rr = ((lo == 0xFF80u) ? 0u : lo) | (((hi16 == 0xFF80u) ? 0u : hi16) << 16);
    *(unsigned*)&s_q[r][c2] = rr;
  }
  __syncthreads();
  {
    v8f pd = boxsum21(&s_q[0][0], &s_h[w][0][0], bl, xl, m16, hi);
    float* conf = out + (size_t)BB * 2 * HH * WW + (size_t)b * HH * WW;
    const int x = x0 + xl + m16;
    #pragma unroll
    for (int j = 0; j < 8; ++j) {
      const int y = y0 + hi * 8 + j;
      float v = pd[j] * (1.0f / 441.0f);
      conf[y * WW + x] = fminf(fmaxf(v, 0.0f), 1.0f);
    }
  }
  __syncthreads();   // conf pass done reading s_q before first offset overwrites it

  // ---- 361-offset argmax scan (dy outer, dx inner: matches reference tie order) ----
  v8f vmax = {-1.f,-1.f,-1.f,-1.f,-1.f,-1.f,-1.f,-1.f};
  v8f bdx  = {0.f,0.f,0.f,0.f,0.f,0.f,0.f,0.f};
  v8f bdy  = {0.f,0.f,0.f,0.f,0.f,0.f,0.f,0.f};

  #pragma unroll 1
  for (int o = 0; o < 361; ++o) {
    const int dy = o / 19 - SRCH;
    const int dx = o % 19 - SRCH;

    // agreement tile: Q = p1*s + (1-p1)*(1-s), 0 outside image (box zero-pad).
    // Processed as column pairs: 1 b32 p1 load, 2 b16 p2 loads, 1 b32 Q store.
    for (int i = tid; i < P1R * (P1C / 2); i += 256) {
      const int r = i / (P1C / 2), c2 = (i % (P1C / 2)) * 2;
      const unsigned u   = *(const unsigned*)&s_p1[r][c2];
      const unsigned p0h = u & 0xFFFFu, p1h = u >> 16;
      const unsigned short* srow = &s_p2[r + SRCH + dy][c2 + SRCH + dx];
      const float s0 = bf2f(srow[0]), s1 = bf2f(srow[1]);
      const float pa = bf2f((unsigned short)p0h), pb = bf2f((unsigned short)p1h);
      float q0 = pa * s0 + (1.0f - pa) * (1.0f - s0);
      float q1 = pb * s1 + (1.0f - pb) * (1.0f - s1);
      q0 = (p0h == 0xFF80u) ? 0.0f : q0;
      q1 = (p1h == 0xFF80u) ? 0.0f : q1;
      *(unsigned*)&s_q[r][c2] = pk2bf(q0, q1);
    }
    __syncthreads();

    const v8f corr = boxsum21(&s_q[0][0], &s_h[w][0][0], bl, xl, m16, hi);
    const float fdx = (float)dx, fdy = (float)dy;
    #pragma unroll
    for (int j = 0; j < 8; ++j) {
      const bool m = corr[j] > vmax[j];
      vmax[j] = m ? corr[j] : vmax[j];
      bdx[j]  = m ? fdx     : bdx[j];
      bdy[j]  = m ? fdy     : bdy[j];
    }
    __syncthreads();   // all waves done reading s_q before next overwrite
  }

  // ---- write flow: channel 0 = dx, channel 1 = dy ----
  float* fx = out + ((size_t)b * 2 + 0) * HH * WW;
  float* fy = out + ((size_t)b * 2 + 1) * HH * WW;
  const int x = x0 + xl + m16;
  #pragma unroll
  for (int j = 0; j < 8; ++j) {
    const int y = y0 + hi * 8 + j;
    fx[y * WW + x] = bdx[j];
    fy[y * WW + x] = bdy[j];
  }
}

extern "C" void kernel_launch(void* const* d_in, const int* in_sizes, int n_in,
                              void* d_out, int out_size, void* d_ws, size_t ws_size,
                              hipStream_t stream) {
  (void)in_sizes; (void)n_in; (void)d_ws; (void)ws_size; (void)out_size;
  const float* p1 = (const float*)d_in[0];
  const float* p2 = (const float*)d_in[1];
  float* out = (float*)d_out;
  dim3 grid(WW / TX, HH / TY, BB);   // 5 x 30 x 4
  BinaryCorrelationMatcher_kernel<<<grid, dim3(256), 0, stream>>>(p1, p2, out);
}
